// AttnDecodingMaxpool_48361331753545
// MI455X (gfx1250) — matrix operand order repaired and checked by reading, not compiled, not run on hardware
//
#include <hip/hip_runtime.h>

// ---------------------------------------------------------------------------
// AttnDecodingMaxpool fused kernel for gfx1250 (MI455X, wave32, WMMA bf16)
//
// Reference: per spatial location p (4096 of them):
//   xs (32x512) -> q = xs@Wq^T+bq, k = xs@Wk^T+bk, v = xs (8 heads, dk=64)
//   attn = softmax(q k^T / 8);  out = attn @ v  (per head)
//   y = out @ Wl^T + bl  (32x20), maxpool over batch -> 20 floats per p
// ---------------------------------------------------------------------------

typedef __attribute__((ext_vector_type(16))) __bf16 v16bf;
typedef __attribute__((ext_vector_type(8)))  __bf16 v8bf;
typedef __attribute__((ext_vector_type(8)))  float  v8f;

#define CCH   512          // channels
#define NB    32           // batch (temporal sequence)
#define HW    4096         // spatial locations
#define NH    8            // heads
#define NCLS  20           // classes
#define XROW  528          // bf16 row stride for 512-wide LDS tiles (16B align)
#define SROW  33           // float row stride for scores / yred
#define AROW  40           // bf16 row stride for attn rows (16B align)

// LDS byte offsets (total 155648 B < 320 KB WGP budget)
#define XS_OFF 0u          // bf16 [32][528]  xs (also V)
#define Q_OFF  33792u      // bf16 [32][528]  q, later aliased by attention out
#define K_OFF  67584u      // bf16 [32][528]  k
#define SC_OFF 101376u     // f32  [8][32][33] scores, later yred [32][33]
#define AT_OFF 135168u     // bf16 [8][32][40] softmax probabilities
#define SMEM_BYTES 155648u

__device__ __forceinline__ v16bf combine8(v8bf lo, v8bf hi) {
  v16bf r;
#pragma unroll
  for (int i = 0; i < 8; ++i) { r[i] = lo[i]; r[i + 8] = hi[i]; }
  return r;
}

// A-fragment (16x32 bf16, row-major source). Contiguous K: lanes 0-15 -> K 0..15,
// lanes 16-31 -> K 16..31 (matches B fragments).
__device__ __forceinline__ v16bf load_a_frag(const __bf16* base, int rowStride,
                                             int mBase, int kBase, int lane) {
  int m = mBase + (lane & 15);
  int s = (lane < 16) ? 0 : 16;
  const __bf16* r = base + m * rowStride + kBase + s;
  return combine8(*(const v8bf*)r, *(const v8bf*)(r + 8));
}

// B-fragment (32x16 bf16) where B[kk][nn] = src[nBase+nn][kBase+kk] (row-major src:
// each lane holds 16 contiguous K values of one source row; lanes 16-31 -> K+16).
__device__ __forceinline__ v16bf load_b_frag_rows(const __bf16* base, int rowStride,
                                                  int nBase, int kBase, int lane) {
  int n  = nBase + (lane & 15);
  int kh = (lane < 16) ? 0 : 16;
  const __bf16* r = base + n * rowStride + kBase + kh;
  return combine8(*(const v8bf*)r, *(const v8bf*)(r + 8));
}

// B-fragment with column-strided source: B[kk][nn] = src[kBase+kk][col(nn)]
__device__ __forceinline__ v16bf load_b_frag_cols(const __bf16* base, int rowStride,
                                                  int kBase, int colBase, int lane) {
  int col = colBase + (lane & 15);
  int kh  = (lane < 16) ? 0 : 16;
  v16bf b;
#pragma unroll
  for (int e = 0; e < 16; ++e) b[e] = base[(kBase + kh + e) * rowStride + col];
  return b;
}

__device__ __forceinline__ v8f wmma_bf16(v16bf a, v16bf b, v8f c) {
  return __builtin_amdgcn_wmma_f32_16x16x32_bf16(false, a, false, b,
                                                 (short)0, c, false, false);
}

// C/D layout: N = lane%16,, M = r + 8*(lane>=16) within a 16x16 tile.
__device__ __forceinline__ void store_c_bf16(__bf16* base, int rowStride, int mBase,
                                             int nBase, v8f acc, float bias, int lane) {
  int col = nBase + (lane & 15);
  int mo  = (lane < 16) ? 0 : 8;
#pragma unroll
  for (int r = 0; r < 8; ++r)
    base[(mBase + mo + r) * rowStride + col] = (__bf16)(acc[r] + bias);
}

__device__ __forceinline__ void store_c_f32(float* base, int rowStride, int mBase,
                                            int nBase, v8f acc, float scale, int lane) {
  int col = nBase + (lane & 15);
  int mo  = (lane < 16) ? 0 : 8;
#pragma unroll
  for (int r = 0; r < 8; ++r)
    base[(mBase + mo + r) * rowStride + col] = acc[r] * scale;
}

// ---------------------------------------------------------------------------
// Prologue: fp32 weights -> bf16 in workspace (stays resident in 192MB L2)
// ---------------------------------------------------------------------------
__global__ void cvt_weights_kernel(const float* __restrict__ Wq,
                                   const float* __restrict__ Wk,
                                   const float* __restrict__ Wl,
                                   __bf16* __restrict__ ws) {
  int i = blockIdx.x * 256 + threadIdx.x;
  const int nqk = CCH * CCH;                 // 262144
  __bf16* wq = ws;
  __bf16* wk = ws + nqk;
  __bf16* wl = ws + 2 * nqk;
  if (i < nqk)                 wq[i]            = (__bf16)Wq[i];
  else if (i < 2 * nqk)        wk[i - nqk]      = (__bf16)Wk[i - nqk];
  else if (i < 2 * nqk + NCLS * CCH) wl[i - 2 * nqk] = (__bf16)Wl[i - 2 * nqk];
}

// ---------------------------------------------------------------------------
// Main fused kernel: one block (8 wave32) per spatial location p
// ---------------------------------------------------------------------------
__global__ void __launch_bounds__(256)
attn_decode_maxpool_kernel(const float* __restrict__ x,
                           const float* __restrict__ bq,
                           const float* __restrict__ bk,
                           const float* __restrict__ bl,
                           const __bf16* __restrict__ wq_bf,
                           const __bf16* __restrict__ wk_bf,
                           const __bf16* __restrict__ wl_bf,
                           float* __restrict__ out) {
  __shared__ alignas(16) unsigned char smem[SMEM_BYTES];
  __bf16* xs = (__bf16*)(smem + XS_OFF);
  __bf16* qm = (__bf16*)(smem + Q_OFF);   // q, then attention-out (same columns/wave)
  __bf16* km = (__bf16*)(smem + K_OFF);
  float*  sc = (float*)(smem + SC_OFF);   // scores, then yred
  __bf16* at = (__bf16*)(smem + AT_OFF);

  const int p    = blockIdx.x;
  const int tid  = threadIdx.x;
  const int lane = tid & 31;
  const int wv   = tid >> 5;

  // ---- Phase 1: gather xs_p (32x512) and convert to bf16 --------------------
  for (int i = tid; i < NB * CCH; i += 256) {
    int n = i >> 9, c = i & 511;
    xs[n * XROW + c] = (__bf16)x[((n << 9) + c) * HW + p];
  }
  __syncthreads();

  // ---- Phase 2: Q and K projections (each wave owns 64 output columns) ------
  {
    const int nB = wv * 64;
#pragma unroll
    for (int pr = 0; pr < 2; ++pr) {
      const __bf16* wgt  = pr ? wk_bf : wq_bf;
      const float*  bias = pr ? bk : bq;
      __bf16*       dst  = pr ? km : qm;
      v8f acc[2][4] = {};
      for (int kt = 0; kt < 16; ++kt) {
        int kB = kt * 32;
        v16bf a0 = load_a_frag(xs, XROW, 0,  kB, lane);
        v16bf a1 = load_a_frag(xs, XROW, 16, kB, lane);
#pragma unroll
        for (int nt = 0; nt < 4; ++nt) {
          v16bf b = load_b_frag_rows(wgt, CCH, nB + nt * 16, kB, lane);
          acc[0][nt] = wmma_bf16(a0, b, acc[0][nt]);
          acc[1][nt] = wmma_bf16(a1, b, acc[1][nt]);
        }
      }
#pragma unroll
      for (int mt = 0; mt < 2; ++mt)
#pragma unroll
        for (int nt = 0; nt < 4; ++nt) {
          float bv = bias[nB + nt * 16 + (lane & 15)];
          store_c_bf16(dst, XROW, mt * 16, nB + nt * 16, acc[mt][nt], bv, lane);
        }
    }
  }
  __syncthreads();

  // ---- Phase 3: attention, wave = head --------------------------------------
  const int hB = wv * 64;
  {
    // scores = q_h k_h^T / 8  -> sc[head][32][33]
    v8f s[2][2] = {};
#pragma unroll
    for (int ks = 0; ks < 2; ++ks) {
      int kB = hB + ks * 32;
      v16bf a0 = load_a_frag(qm, XROW, 0,  kB, lane);
      v16bf a1 = load_a_frag(qm, XROW, 16, kB, lane);
#pragma unroll
      for (int nt = 0; nt < 2; ++nt) {
        v16bf b = load_b_frag_rows(km, XROW, nt * 16, kB, lane);
        s[0][nt] = wmma_bf16(a0, b, s[0][nt]);
        s[1][nt] = wmma_bf16(a1, b, s[1][nt]);
      }
    }
    float* scw = sc + wv * NB * SROW;
#pragma unroll
    for (int mt = 0; mt < 2; ++mt)
#pragma unroll
      for (int nt = 0; nt < 2; ++nt)
        store_c_f32(scw, SROW, mt * 16, nt * 16, s[mt][nt], 0.125f, lane);
  }
  __syncthreads();

  // softmax: each lane owns one row (query n = lane) of its head's 32x32 scores
  {
    float* scw = sc + wv * NB * SROW;
    const float* row = scw + lane * SROW;
    float mx = -3.0e38f;
#pragma unroll
    for (int m = 0; m < 32; ++m) mx = fmaxf(mx, row[m]);
    float e[32], sum = 0.0f;
#pragma unroll
    for (int m = 0; m < 32; ++m) { e[m] = __expf(row[m] - mx); sum += e[m]; }
    float inv = 1.0f / sum;
    __bf16* arow = at + wv * NB * AROW + lane * AROW;
#pragma unroll
    for (int m = 0; m < 32; ++m) arow[m] = (__bf16)(e[m] * inv);
  }
  __syncthreads();

  // out_h = attn (32x32) @ v_h (32x64); written over qm (wave-private columns)
  {
    const __bf16* ah = at + wv * NB * AROW;
    v16bf a0 = load_a_frag(ah, AROW, 0,  0, lane);
    v16bf a1 = load_a_frag(ah, AROW, 16, 0, lane);
#pragma unroll
    for (int dt = 0; dt < 4; ++dt) {
      v16bf b = load_b_frag_cols(xs, XROW, 0, hB + dt * 16, lane);
      v8f o0 = {}, o1 = {};
      o0 = wmma_bf16(a0, b, o0);
      o1 = wmma_bf16(a1, b, o1);
      store_c_bf16(qm, XROW, 0,  hB + dt * 16, o0, 0.0f, lane);
      store_c_bf16(qm, XROW, 16, hB + dt * 16, o1, 0.0f, lane);
    }
  }
  __syncthreads();

  // ---- Phase 4: classifier (32x512)@(512x20pad32) on waves 0..3 -------------
  if (wv < 4) {
    const int mt = wv & 1, nt = wv >> 1;
    const int g  = nt * 16 + (lane & 15);   // class row for this lane
    const int kh = (lane < 16) ? 0 : 16;
    v8f yacc = {};
    for (int kt = 0; kt < 16; ++kt) {
      int kB = kt * 32;
      v16bf a = load_a_frag(qm, XROW, mt * 16, kB, lane);
      v16bf b;
      if (g < NCLS) {
        const __bf16* r = wl_bf + g * CCH + kB + kh;
        b = combine8(*(const v8bf*)r, *(const v8bf*)(r + 8));
      } else {
#pragma unroll
        for (int e = 0; e < 16; ++e) b[e] = (__bf16)0.0f;
      }
      yacc = wmma_bf16(a, b, yacc);
    }
    // yred[32 batch][32 cls-padded] into sc region
    int col = nt * 16 + (lane & 15);
    int mo  = (lane < 16) ? 0 : 8;
#pragma unroll
    for (int r = 0; r < 8; ++r)
      sc[(mt * 16 + mo + r) * SROW + col] = yacc[r];
  }
  __syncthreads();

  // max over batch, add bias, write 20 outputs for this p
  if (tid < NCLS) {
    float mx = -3.0e38f;
#pragma unroll
    for (int n = 0; n < NB; ++n) mx = fmaxf(mx, sc[n * SROW + tid]);
    out[p * NCLS + tid] = mx + bl[tid];
  }
}

// ---------------------------------------------------------------------------
extern "C" void kernel_launch(void* const* d_in, const int* in_sizes, int n_in,
                              void* d_out, int out_size, void* d_ws, size_t ws_size,
                              hipStream_t stream) {
  const float* x  = (const float*)d_in[0];
  const float* Wq = (const float*)d_in[1];
  const float* bq = (const float*)d_in[2];
  const float* Wk = (const float*)d_in[3];
  const float* bk = (const float*)d_in[4];
  const float* Wl = (const float*)d_in[5];
  const float* bl = (const float*)d_in[6];
  float* out = (float*)d_out;

  __bf16* ws = (__bf16*)d_ws;              // [Wq_bf | Wk_bf | Wl_bf]
  const int nqk = CCH * CCH;
  const int total = 2 * nqk + NCLS * CCH;  // 534528 elements

  cvt_weights_kernel<<<(total + 255) / 256, 256, 0, stream>>>(Wq, Wk, Wl, ws);

  const __bf16* wq_bf = ws;
  const __bf16* wk_bf = ws + nqk;
  const __bf16* wl_bf = ws + 2 * nqk;
  attn_decode_maxpool_kernel<<<HW, 256, 0, stream>>>(x, bq, bk, bl,
                                                     wq_bf, wk_bf, wl_bf, out);
}
